// CUTLASSGroupedLinear_9363028706407
// MI455X (gfx1250) — matrix operand order repaired and checked by reading, compiled, not run on hardware
//
#include <hip/hip_runtime.h>

typedef __attribute__((ext_vector_type(16))) __bf16 v16bf;
typedef __attribute__((ext_vector_type(8)))  __bf16 v8bf;
typedef __attribute__((ext_vector_type(4)))  __bf16 v4bf;
typedef __attribute__((ext_vector_type(8)))  float  v8f;
typedef __attribute__((ext_vector_type(4)))  unsigned short v4us;
typedef __attribute__((ext_vector_type(8)))  unsigned short v8us;
typedef int vi4 __attribute__((vector_size(16)));

#define NUM_E   8
#define K_DIM   1024
#define N_DIM   2048
#define T_TOT   16384
#define BM      128
#define BN      128
#define BK      32
#define LDA     40            // bf16 elems per LDS row (32 + 8 pad): 80B rows, conflict-free, 16B aligned
#define NKSTEP  (K_DIM / BK)
#define PLANE   (BM * LDA)    // one LDS plane (rows x padded K), in bf16 elems

#if defined(__HIP_DEVICE_COMPILE__)
#  if __has_builtin(__builtin_amdgcn_global_load_async_to_lds_b128) && \
      __has_builtin(__builtin_amdgcn_s_wait_asynccnt)
#    define HAVE_ASYNC 1
#  else
#    define HAVE_ASYNC 0
#  endif
#else
#  define HAVE_ASYNC 0
#endif

// builtin wants (int4 AS1*, int4 AS3*, imm offset, imm cpol) per the round-2 diagnostic
#define GP1(p) ((__attribute__((address_space(1))) vi4*)(void*)(p))
#define LP3(p) ((__attribute__((address_space(3))) vi4*)(void*)(p))

__device__ __forceinline__ __bf16 bft(float f) {
    unsigned u = __builtin_bit_cast(unsigned, f);
    unsigned short h = (unsigned short)(u >> 16);   // truncate to bf16
    return __builtin_bit_cast(__bf16, h);
}
__device__ __forceinline__ float bff(__bf16 b) {
    unsigned short h = __builtin_bit_cast(unsigned short, b);
    unsigned u = ((unsigned)h) << 16;
    return __builtin_bit_cast(float, u);
}

__global__ __launch_bounds__(256) void zero_f4(float4* __restrict__ o, long n4) {
    long i = (long)blockIdx.x * blockDim.x + threadIdx.x;
    if (i < n4) o[i] = make_float4(0.f, 0.f, 0.f, 0.f);
}

// One-shot pre-pass: split f32 into bf16 hi/lo planes (hi = trunc(x), lo = trunc(x - hi)).
__global__ __launch_bounds__(256) void split_bf16(const float* __restrict__ src,
                                                  unsigned short* __restrict__ hi,
                                                  unsigned short* __restrict__ lo, long n4) {
    long i = (long)blockIdx.x * blockDim.x + threadIdx.x;
    if (i >= n4) return;
    float4 f = ((const float4*)src)[i];
    float v[4] = {f.x, f.y, f.z, f.w};
    v4us h, l;
#pragma unroll
    for (int j = 0; j < 4; ++j) {
        unsigned u = __builtin_bit_cast(unsigned, v[j]);
        h[j] = (unsigned short)(u >> 16);
        float hf = __builtin_bit_cast(float, u & 0xffff0000u);
        l[j] = (unsigned short)(__builtin_bit_cast(unsigned, v[j] - hf) >> 16);
    }
    ((v4us*)hi)[i] = h;
    ((v4us*)lo)[i] = l;
}

// ---------------------------------------------------------------------------
// Main path: bf16 planes already in global memory; async-copy tiles to LDS,
// double-buffered; 24 v_wmma_f32_16x16x32_bf16 per wave per K-step (bf16x3).
// ---------------------------------------------------------------------------
__global__ __launch_bounds__(256) void grouped_gemm_planes(
    const unsigned short* __restrict__ Xhi, const unsigned short* __restrict__ Xlo,
    const unsigned short* __restrict__ Whi, const unsigned short* __restrict__ Wlo,
    const int* __restrict__ m_sizes, const int* __restrict__ m_offsets,
    float* __restrict__ Y)
{
    extern __shared__ unsigned short sdyn[];   // [2 bufs][4 planes][PLANE] = 80KB

    const int e     = blockIdx.z;
    const int msize = m_sizes[e];
    const int moff  = m_offsets[e];
    const int mBase = blockIdx.x * BM;
    if (mBase >= msize) return;
    const int nBase = blockIdx.y * BN;

    const int tid   = threadIdx.x;
    const int lane  = tid & 31;
    const int wave  = tid >> 5;
    const int waveM = wave >> 2;   // 0..1
    const int waveN = wave & 3;    // 0..3
    const int lr    = lane & 15;
    const int hlf   = lane >> 4;

    const size_t wOff = (size_t)e * N_DIM * K_DIM;

    // Staging map: each plane is 128 rows x 32 bf16 = 512 16B-chunks.
    // wave w moves chunks {w*64 + i*32 + lane : i in 0..1} of every plane -> 8 async ops/wave/tile.
    int rowC[2], colC[2], mrow[2], nrow[2];
#pragma unroll
    for (int i = 0; i < 2; ++i) {
        int c   = wave * 64 + i * 32 + lane;
        rowC[i] = c >> 2;
        colC[i] = (c & 3) * 8;                       // bf16 elems within row
        int mr  = mBase + rowC[i];
        mrow[i] = moff + (mr < msize ? mr : msize - 1);  // clamp: stores are guarded later
        nrow[i] = nBase + rowC[i];
    }

    auto issue = [&](int kt, int b) {
        const int kb = kt * BK;
        unsigned short* base = sdyn + (size_t)b * 4 * PLANE;
#pragma unroll
        for (int i = 0; i < 2; ++i) {
            const unsigned short* sA0 = Xhi + (size_t)mrow[i] * K_DIM + kb + colC[i];
            const unsigned short* sA1 = Xlo + (size_t)mrow[i] * K_DIM + kb + colC[i];
            const unsigned short* sB0 = Whi + wOff + (size_t)nrow[i] * K_DIM + kb + colC[i];
            const unsigned short* sB1 = Wlo + wOff + (size_t)nrow[i] * K_DIM + kb + colC[i];
            unsigned short* dA0 = base + 0 * PLANE + rowC[i] * LDA + colC[i];
            unsigned short* dA1 = base + 1 * PLANE + rowC[i] * LDA + colC[i];
            unsigned short* dB0 = base + 2 * PLANE + rowC[i] * LDA + colC[i];
            unsigned short* dB1 = base + 3 * PLANE + rowC[i] * LDA + colC[i];
#if HAVE_ASYNC
            __builtin_amdgcn_global_load_async_to_lds_b128(GP1(sA0), LP3(dA0), 0, 0);
            __builtin_amdgcn_global_load_async_to_lds_b128(GP1(sA1), LP3(dA1), 0, 0);
            __builtin_amdgcn_global_load_async_to_lds_b128(GP1(sB0), LP3(dB0), 0, 0);
            __builtin_amdgcn_global_load_async_to_lds_b128(GP1(sB1), LP3(dB1), 0, 0);
#else
            *(v8us*)dA0 = *(const v8us*)sA0;
            *(v8us*)dA1 = *(const v8us*)sA1;
            *(v8us*)dB0 = *(const v8us*)sB0;
            *(v8us*)dB1 = *(const v8us*)sB1;
#endif
        }
    };

    v8f acc[4][2];
#pragma unroll
    for (int i = 0; i < 4; ++i)
#pragma unroll
        for (int j = 0; j < 2; ++j)
#pragma unroll
            for (int k = 0; k < 8; ++k) acc[i][j][k] = 0.f;

    issue(0, 0);

    for (int kt = 0; kt < NKSTEP; ++kt) {
        const int  b       = kt & 1;
        const bool hasNext = (kt + 1 < NKSTEP);
        // Trailing barrier of previous iteration guarantees nobody still reads buf[1-b].
        if (hasNext) issue(kt + 1, 1 - b);
#if HAVE_ASYNC
        if (hasNext) __builtin_amdgcn_s_wait_asynccnt(8);   // tile kt done; kt+1 (8 ops) may fly
        else         __builtin_amdgcn_s_wait_asynccnt(0);
#endif
        __syncthreads();                                    // all waves' tile-kt data visible

        const unsigned short* sb = sdyn + (size_t)b * 4 * PLANE;

        // A fragments (16-bit A 16x32 layout: lanes 0-15 K=0..7,16..23; lanes 16-31 K=8..15,24..31)
        v16bf ahi[4], alo[4];
#pragma unroll
        for (int mt = 0; mt < 4; ++mt) {
            const int o = (waveM * 64 + mt * 16 + lr) * LDA + hlf * 8;
            v8bf h0 = *(const v8bf*)(const void*)(sb + 0 * PLANE + o);
            v8bf h1 = *(const v8bf*)(const void*)(sb + 0 * PLANE + o + 16);
            ahi[mt] = __builtin_shufflevector(h0, h1, 0,1,2,3,4,5,6,7,8,9,10,11,12,13,14,15);
            v8bf l0 = *(const v8bf*)(const void*)(sb + 1 * PLANE + o);
            v8bf l1 = *(const v8bf*)(const void*)(sb + 1 * PLANE + o + 16);
            alo[mt] = __builtin_shufflevector(l0, l1, 0,1,2,3,4,5,6,7,8,9,10,11,12,13,14,15);
        }
        // B fragments (32x16: lane = column, lanes 0-15 K=0..7,16..23; lanes 16-31 K=8..15,24..31)
        v16bf bhi[2], blo[2];
#pragma unroll
        for (int nt = 0; nt < 2; ++nt) {
            const int o = (waveN * 32 + nt * 16 + lr) * LDA + hlf * 8;
            v8bf h0 = *(const v8bf*)(const void*)(sb + 2 * PLANE + o);
            v8bf h1 = *(const v8bf*)(const void*)(sb + 2 * PLANE + o + 16);
            bhi[nt] = __builtin_shufflevector(h0, h1, 0,1,2,3,4,5,6,7,8,9,10,11,12,13,14,15);
            v8bf l0 = *(const v8bf*)(const void*)(sb + 3 * PLANE + o);
            v8bf l1 = *(const v8bf*)(const void*)(sb + 3 * PLANE + o + 16);
            blo[nt] = __builtin_shufflevector(l0, l1, 0,1,2,3,4,5,6,7,8,9,10,11,12,13,14,15);
        }

#pragma unroll
        for (int mt = 0; mt < 4; ++mt)
#pragma unroll
            for (int nt = 0; nt < 2; ++nt) {
                acc[mt][nt] = __builtin_amdgcn_wmma_f32_16x16x32_bf16(
                    false, ahi[mt], false, bhi[nt], (short)0, acc[mt][nt], false, false);
                acc[mt][nt] = __builtin_amdgcn_wmma_f32_16x16x32_bf16(
                    false, ahi[mt], false, blo[nt], (short)0, acc[mt][nt], false, false);
                acc[mt][nt] = __builtin_amdgcn_wmma_f32_16x16x32_bf16(
                    false, alo[mt], false, bhi[nt], (short)0, acc[mt][nt], false, false);
            }

        __syncthreads();   // all waves done reading buf[b] before it is refilled
    }

    // Epilogue. C/D layout: VGPR r -> row (r + 8*hlf), col = lane&15.
#pragma unroll
    for (int mt = 0; mt < 4; ++mt) {
#pragma unroll
        for (int r = 0; r < 8; ++r) {
            const int ml = waveM * 64 + mt * 16 + hlf * 8 + r;
            const int m  = mBase + ml;
            if (m < msize) {
                float* yrow = Y + (size_t)(moff + m) * N_DIM + nBase + waveN * 32 + lr;
#pragma unroll
                for (int nt = 0; nt < 2; ++nt)
                    yrow[nt * 16] = acc[mt][nt][r];
            }
        }
    }
}

// ---------------------------------------------------------------------------
// Fallback (ws too small): fused conversion-in-LDS kernel (round-1 version).
// ---------------------------------------------------------------------------
__global__ __launch_bounds__(256) void grouped_gemm_bf16x3(
    const float* __restrict__ X, const float* __restrict__ W,
    const int* __restrict__ m_sizes, const int* __restrict__ m_offsets,
    float* __restrict__ Y)
{
    __shared__ __bf16 sAhi[BM * LDA];
    __shared__ __bf16 sAlo[BM * LDA];
    __shared__ __bf16 sBhi[BN * LDA];
    __shared__ __bf16 sBlo[BN * LDA];

    const int e     = blockIdx.z;
    const int msize = m_sizes[e];
    const int moff  = m_offsets[e];
    const int mBase = blockIdx.x * BM;
    if (mBase >= msize) return;
    const int nBase = blockIdx.y * BN;

    const int tid   = threadIdx.x;
    const int lane  = tid & 31;
    const int wave  = tid >> 5;
    const int waveM = wave >> 2;
    const int waveN = wave & 3;
    const int lr    = lane & 15;
    const int hlf   = lane >> 4;

    const float* Wbase = W + (size_t)e * N_DIM * K_DIM;

    int rowS[4], colS[4];
#pragma unroll
    for (int p = 0; p < 4; ++p) {
        int lin = p * 256 + tid;
        rowS[p] = lin >> 3;
        colS[p] = (lin & 7) * 4;
    }

    float4 pa[4], pb[4];

    auto gload = [&](int kt) {
        const int kBase = kt * BK;
#pragma unroll
        for (int p = 0; p < 4; ++p) {
            int m = mBase + rowS[p];
            if (m < msize)
                pa[p] = *(const float4*)(X + (size_t)(moff + m) * K_DIM + kBase + colS[p]);
            else
                pa[p] = make_float4(0.f, 0.f, 0.f, 0.f);
            pb[p] = *(const float4*)(Wbase + (size_t)(nBase + rowS[p]) * K_DIM + kBase + colS[p]);
        }
    };

    auto sstore = [&]() {
#pragma unroll
        for (int p = 0; p < 4; ++p) {
            float fa[4] = {pa[p].x, pa[p].y, pa[p].z, pa[p].w};
            float fb[4] = {pb[p].x, pb[p].y, pb[p].z, pb[p].w};
            v4bf ah, al, bh, bl;
#pragma unroll
            for (int j = 0; j < 4; ++j) {
                __bf16 h = bft(fa[j]); ah[j] = h; al[j] = bft(fa[j] - bff(h));
                __bf16 g = bft(fb[j]); bh[j] = g; bl[j] = bft(fb[j] - bff(g));
            }
            int o = rowS[p] * LDA + colS[p];
            *(v4bf*)(sAhi + o) = ah;  *(v4bf*)(sAlo + o) = al;
            *(v4bf*)(sBhi + o) = bh;  *(v4bf*)(sBlo + o) = bl;
        }
    };

    v8f acc[4][2];
#pragma unroll
    for (int i = 0; i < 4; ++i)
#pragma unroll
        for (int j = 0; j < 2; ++j)
#pragma unroll
            for (int k = 0; k < 8; ++k) acc[i][j][k] = 0.f;

    gload(0);
    sstore();
    __syncthreads();

    for (int kt = 0; kt < NKSTEP; ++kt) {
        const bool hasNext = (kt + 1 < NKSTEP);
        if (hasNext) gload(kt + 1);

        v16bf ahi[4], alo[4];
#pragma unroll
        for (int mt = 0; mt < 4; ++mt) {
            const int base = (waveM * 64 + mt * 16 + lr) * LDA + hlf * 8;
            v8bf h0 = *(const v8bf*)(sAhi + base);
            v8bf h1 = *(const v8bf*)(sAhi + base + 16);
            ahi[mt] = __builtin_shufflevector(h0, h1, 0,1,2,3,4,5,6,7,8,9,10,11,12,13,14,15);
            v8bf l0 = *(const v8bf*)(sAlo + base);
            v8bf l1 = *(const v8bf*)(sAlo + base + 16);
            alo[mt] = __builtin_shufflevector(l0, l1, 0,1,2,3,4,5,6,7,8,9,10,11,12,13,14,15);
        }
        v16bf bhi[2], blo[2];
#pragma unroll
        for (int nt = 0; nt < 2; ++nt) {
            const int base = (waveN * 32 + nt * 16 + lr) * LDA + hlf * 8;
            v8bf h0 = *(const v8bf*)(sBhi + base);
            v8bf h1 = *(const v8bf*)(sBhi + base + 16);
            bhi[nt] = __builtin_shufflevector(h0, h1, 0,1,2,3,4,5,6,7,8,9,10,11,12,13,14,15);
            v8bf l0 = *(const v8bf*)(sBlo + base);
            v8bf l1 = *(const v8bf*)(sBlo + base + 16);
            blo[nt] = __builtin_shufflevector(l0, l1, 0,1,2,3,4,5,6,7,8,9,10,11,12,13,14,15);
        }

#pragma unroll
        for (int mt = 0; mt < 4; ++mt)
#pragma unroll
            for (int nt = 0; nt < 2; ++nt) {
                acc[mt][nt] = __builtin_amdgcn_wmma_f32_16x16x32_bf16(
                    false, ahi[mt], false, bhi[nt], (short)0, acc[mt][nt], false, false);
                acc[mt][nt] = __builtin_amdgcn_wmma_f32_16x16x32_bf16(
                    false, ahi[mt], false, blo[nt], (short)0, acc[mt][nt], false, false);
                acc[mt][nt] = __builtin_amdgcn_wmma_f32_16x16x32_bf16(
                    false, alo[mt], false, bhi[nt], (short)0, acc[mt][nt], false, false);
            }

        __syncthreads();
        if (hasNext) {
            sstore();
            __syncthreads();
        }
    }

#pragma unroll
    for (int mt = 0; mt < 4; ++mt) {
#pragma unroll
        for (int r = 0; r < 8; ++r) {
            const int ml = waveM * 64 + mt * 16 + hlf * 8 + r;
            const int m  = mBase + ml;
            if (m < msize) {
                float* yrow = Y + (size_t)(moff + m) * N_DIM + nBase + waveN * 32 + lr;
#pragma unroll
                for (int nt = 0; nt < 2; ++nt)
                    yrow[nt * 16] = acc[mt][nt][r];
            }
        }
    }
}

extern "C" void kernel_launch(void* const* d_in, const int* in_sizes, int n_in,
                              void* d_out, int out_size, void* d_ws, size_t ws_size,
                              hipStream_t stream) {
    (void)in_sizes; (void)n_in;
    const float* X  = (const float*)d_in[0];
    const float* W  = (const float*)d_in[1];
    const int*   ms = (const int*)d_in[2];
    const int*   mo = (const int*)d_in[3];
    float*       Y  = (float*)d_out;

    // Clear output (harness poisons it; tokens outside all groups must be zero).
    long n4 = (long)out_size / 4;
    zero_f4<<<dim3((unsigned)((n4 + 255) / 256)), dim3(256), 0, stream>>>((float4*)d_out, n4);

    dim3 grid(T_TOT / BM, N_DIM / BN, NUM_E);

    const size_t planeX = (size_t)T_TOT * K_DIM;            // elems
    const size_t planeW = (size_t)NUM_E * N_DIM * K_DIM;    // elems
    const size_t need   = 2 * (planeX + planeW) * sizeof(unsigned short);  // 128 MiB

    if (ws_size >= need) {
        unsigned short* xhi = (unsigned short*)d_ws;
        unsigned short* xlo = xhi + planeX;
        unsigned short* whi = xlo + planeX;
        unsigned short* wlo = whi + planeW;
        long nx4 = (long)(planeX / 4);
        long nw4 = (long)(planeW / 4);
        split_bf16<<<dim3((unsigned)((nx4 + 255) / 256)), dim3(256), 0, stream>>>(X, xhi, xlo, nx4);
        split_bf16<<<dim3((unsigned)((nw4 + 255) / 256)), dim3(256), 0, stream>>>(W, whi, wlo, nw4);
        size_t ldsBytes = (size_t)2 * 4 * PLANE * sizeof(unsigned short);  // 80 KB dynamic LDS
        grouped_gemm_planes<<<grid, dim3(256), ldsBytes, stream>>>(xhi, xlo, whi, wlo, ms, mo, Y);
    } else {
        grouped_gemm_bf16x3<<<grid, dim3(256), 0, stream>>>(X, W, ms, mo, Y);
    }
}
